// DFloat11Linear_206158430637
// MI455X (gfx1250) — compile-verified, matrix operand order checked by reading, not run on hardware
//
#include <hip/hip_runtime.h>

// ---- CDNA5 (gfx1250) WMMA types -------------------------------------------
typedef __attribute__((ext_vector_type(16))) __bf16 v16bf;
typedef __attribute__((ext_vector_type(8)))  float  v8f;

#define IN_F   4096
#define OUT_F  11008
#define M_TOT  4096

#define BM 256        // block tile M (x rows)
#define BN 128        // block tile N (w rows)
#define BK 32         // K slab (one bf16 WMMA K)
#define NSLAB (IN_F / BK)   // 128 (even)
#define LDSK 40       // padded LDS row stride (elements): 80B rows, conflict-free

// pack two f32 -> two bf16 (round-to-nearest, ties-away) in ONE v_perm_b32
__device__ __forceinline__ unsigned f32x2_to_bf16x2(float lo, float hi) {
  unsigned ul = __builtin_bit_cast(unsigned, lo) + 0x8000u;
  unsigned uh = __builtin_bit_cast(unsigned, hi) + 0x8000u;
  return __builtin_amdgcn_perm(uh, ul, 0x07060302u);   // {uh[31:16], ul[31:16]}
}

// decode two DFloat11 weights at once in packed 16-bit lanes
__device__ __forceinline__ unsigned decode2(int s0, int e0, int s1, int e1) {
  unsigned sp = __builtin_amdgcn_perm((unsigned)s1, (unsigned)s0, 0x03040100u);
  unsigned ep = __builtin_amdgcn_perm((unsigned)e1, (unsigned)e0, 0x03040100u);
  unsigned sgn  = (sp & 0x00800080u) << 8;   // sign -> bits 15 / 31
  unsigned expb = (ep & 0x00FF00FFu) << 7;   // exp  -> bits 14:7 / 30:23
  unsigned man  =  sp & 0x007F007Fu;         // mant -> bits  6:0 / 22:16
  return sgn | expb | man;
}

struct XStage { float4 v[8]; };            // one x row: 32 f32
struct WStage { int4 s[4]; int4 e[4]; };   // 16 weights (sm + exp)

__device__ __forceinline__ void load_x(const float* __restrict__ src, XStage& xs) {
  #pragma unroll
  for (int j = 0; j < 8; ++j) xs.v[j] = *(const float4*)(src + j * 4);
}

__device__ __forceinline__ void load_w(const int* __restrict__ sm, const int* __restrict__ ex,
                                       size_t idx, WStage& ws) {
  #pragma unroll
  for (int j = 0; j < 4; ++j) {
    ws.s[j] = *(const int4*)(sm + idx + j * 4);
    ws.e[j] = *(const int4*)(ex + idx + j * 4);
  }
}

__device__ __forceinline__ void store_x(const XStage& xs, unsigned short* dst) {
  #pragma unroll
  for (int j = 0; j < 4; ++j) {
    uint4 p;
    p.x = f32x2_to_bf16x2(xs.v[2 * j].x,     xs.v[2 * j].y);
    p.y = f32x2_to_bf16x2(xs.v[2 * j].z,     xs.v[2 * j].w);
    p.z = f32x2_to_bf16x2(xs.v[2 * j + 1].x, xs.v[2 * j + 1].y);
    p.w = f32x2_to_bf16x2(xs.v[2 * j + 1].z, xs.v[2 * j + 1].w);
    *(uint4*)(dst + j * 8) = p;
  }
}

__device__ __forceinline__ void store_w(const WStage& ws, unsigned short* dst) {
  #pragma unroll
  for (int j = 0; j < 2; ++j) {
    uint4 p;
    p.x = decode2(ws.s[2 * j].x,     ws.e[2 * j].x,     ws.s[2 * j].y,     ws.e[2 * j].y);
    p.y = decode2(ws.s[2 * j].z,     ws.e[2 * j].z,     ws.s[2 * j].w,     ws.e[2 * j].w);
    p.z = decode2(ws.s[2 * j + 1].x, ws.e[2 * j + 1].x, ws.s[2 * j + 1].y, ws.e[2 * j + 1].y);
    p.w = decode2(ws.s[2 * j + 1].z, ws.e[2 * j + 1].z, ws.s[2 * j + 1].w, ws.e[2 * j + 1].w);
    *(uint4*)(dst + j * 8) = p;
  }
}

union Frag { v16bf v; uint4 q[2]; };

// 64x64 wave tile: 4 A frags (cached), B frags streamed -> 16 WMMA / 16 ds_load
__device__ __forceinline__ void compute_tile(const unsigned short* __restrict__ sxb,
                                             const unsigned short* __restrict__ swb,
                                             int wm, int wn, int r, int hl,
                                             v8f acc[4][4]) {
  Frag a[4];
  #pragma unroll
  for (int mi = 0; mi < 4; ++mi) {
    const unsigned short* p = sxb + (wm + mi * 16 + r) * LDSK + hl * 8;
    a[mi].q[0] = *(const uint4*)(p);
    a[mi].q[1] = *(const uint4*)(p + 16);
  }
  #pragma unroll
  for (int ni = 0; ni < 4; ++ni) {
    Frag b;
    const unsigned short* p = swb + (wn + ni * 16 + r) * LDSK + hl * 8;
    b.q[0] = *(const uint4*)(p);
    b.q[1] = *(const uint4*)(p + 16);
    #pragma unroll
    for (int mi = 0; mi < 4; ++mi)
      acc[mi][ni] = __builtin_amdgcn_wmma_f32_16x16x32_bf16(
          false, a[mi].v, false, b.v, (short)0, acc[mi][ni], false, false);
  }
}

__global__ __launch_bounds__(256, 1)
void dfloat11_linear_wmma(const float* __restrict__ x,
                          const int*   __restrict__ smant,
                          const int*   __restrict__ expo,
                          float*       __restrict__ y)
{
  __shared__ unsigned short sx[2][BM * LDSK];   // 2 x 20 KB
  __shared__ unsigned short sw[2][BN * LDSK];   // 2 x 10 KB

  const int t    = threadIdx.x;
  const int lane = t & 31;
  const int wave = t >> 5;
  const int r    = lane & 15;
  const int hl   = lane >> 4;

  const int m0 = blockIdx.y * BM;
  const int n0 = blockIdx.x * BN;
  const int wm = (wave & 3) * 64;     // 4 M-waves
  const int wn = (wave >> 2) * 64;    // 2 N-waves

  // per-thread staging coordinates (constant across slabs)
  const float* xg   = x + (size_t)(m0 + t) * IN_F;             // x row t
  const int    wrow = t >> 1;                                  // w row, half each
  const int    wcol = (t & 1) * 16;
  const size_t widx = (size_t)(n0 + wrow) * IN_F + wcol;

  unsigned short* const sxd0 = &sx[0][t * LDSK];
  unsigned short* const sxd1 = &sx[1][t * LDSK];
  unsigned short* const swd0 = &sw[0][wrow * LDSK + wcol];
  unsigned short* const swd1 = &sw[1][wrow * LDSK + wcol];

  v8f acc[4][4];
  #pragma unroll
  for (int mi = 0; mi < 4; ++mi)
    #pragma unroll
    for (int ni = 0; ni < 4; ++ni) {
      v8f z = {};
      acc[mi][ni] = z;
    }

  XStage xs;
  WStage ws;

  // ---- prologue: stage slab 0 into buffer 0 -------------------------------
  load_x(xg, xs);
  load_w(smant, expo, widx, ws);
  store_x(xs, sxd0);
  store_w(ws, swd0);
  __syncthreads();

  // ---- main loop: 2 slabs/iter, hard-coded ping-pong, branch-free body ----
  #pragma unroll 1
  for (int s = 0; s < NSLAB - 2; s += 2) {
    // slab s (buf0) computes; slab s+1 staged into buf1
    load_x(xg + (s + 1) * BK, xs);
    load_w(smant, expo, widx + (size_t)(s + 1) * BK, ws);
    compute_tile(&sx[0][0], &sw[0][0], wm, wn, r, hl, acc);
    store_x(xs, sxd1);
    store_w(ws, swd1);
    __syncthreads();

    // slab s+1 (buf1) computes; slab s+2 staged into buf0
    load_x(xg + (s + 2) * BK, xs);
    load_w(smant, expo, widx + (size_t)(s + 2) * BK, ws);
    compute_tile(&sx[1][0], &sw[1][0], wm, wn, r, hl, acc);
    store_x(xs, sxd0);
    store_w(ws, swd0);
    __syncthreads();
  }

  // ---- tail: slabs NSLAB-2 (buf0) and NSLAB-1 (buf1) ----------------------
  load_x(xg + (NSLAB - 1) * BK, xs);
  load_w(smant, expo, widx + (size_t)(NSLAB - 1) * BK, ws);
  compute_tile(&sx[0][0], &sw[0][0], wm, wn, r, hl, acc);
  store_x(xs, sxd1);
  store_w(ws, swd1);
  __syncthreads();
  compute_tile(&sx[1][0], &sw[1][0], wm, wn, r, hl, acc);

  // ---- store C: VGPR i -> M = base + hl*8 + i, N = base + r ---------------
  #pragma unroll
  for (int mi = 0; mi < 4; ++mi)
    #pragma unroll
    for (int ni = 0; ni < 4; ++ni) {
      const int gn = n0 + wn + ni * 16 + r;
      #pragma unroll
      for (int i = 0; i < 8; ++i) {
        const int gm = m0 + wm + mi * 16 + hl * 8 + i;
        y[(size_t)gm * OUT_F + gn] = acc[mi][ni][i];
      }
    }
}

extern "C" void kernel_launch(void* const* d_in, const int* in_sizes, int n_in,
                              void* d_out, int out_size, void* d_ws, size_t ws_size,
                              hipStream_t stream) {
  const float* x     = (const float*)d_in[0];
  const int*   smant = (const int*)d_in[1];
  const int*   expo  = (const int*)d_in[2];
  float*       y     = (float*)d_out;

  dim3 grid(OUT_F / BN, M_TOT / BM);    // 86 x 16
  dim3 block(256);
  dfloat11_linear_wmma<<<grid, block, 0, stream>>>(x, smant, expo, y);
}
